// FirstAttn_16234976378891
// MI455X (gfx1250) — matrix operand order repaired and checked by reading, not compiled, not run on hardware
//
#include <hip/hip_runtime.h>
#include <hip/hip_bf16.h>

// ---------------------------------------------------------------------------
// FirstAttn for MI455X (gfx1250, wave32, WMMA).
//
// Shapes: B=1, Cp=Cl=32, D=128, H=W=256, WS=16, NHEADS=8, head_dim=16.
// Window partition in the reference is DILATED: window (wy,wx) holds pixels
// (h = iy*16+wy, w = ix*16+wx), m = iy*16+ix.
//
// Pipeline:
//   K1 ln_stats   : per-pixel mean/rstd over 32 channels (pan & lms)
//   K2 conv1x1_ln : fused LN + 1x1 conv (GEMM 128/256 x 32 x 65536), f16 out
//   K3 dw3x3      : depthwise 3x3 SAME, f16 out
//   K4 attn       : per (window,head): k-normalize, S=q^T k via
//                   v_wmma_f32_16x16x32_f16 (K=16 zero-padded), softmax
//                   (row max+sum fused in LDS pass; attn f32 -> d_out,
//                   coalesced), out = P v^T via WMMA (K=256, 8 steps).
// ---------------------------------------------------------------------------

typedef _Float16 v16h __attribute__((ext_vector_type(16)));
typedef _Float16 v8h  __attribute__((ext_vector_type(8)));
typedef float    v8f  __attribute__((ext_vector_type(8)));

union V16 { v16h v; v8h h[2]; };

__device__ inline v8h v8h_zero() {
  v8h z;
  #pragma unroll
  for (int i = 0; i < 8; ++i) z[i] = (_Float16)0.0f;
  return z;
}
__device__ inline v8f v8f_zero() {
  v8f z;
  #pragma unroll
  for (int i = 0; i < 8; ++i) z[i] = 0.0f;
  return z;
}

#define HW    65536
#define QP    24      // qT/kT row pitch in halves ([pos][d], 16 used) -> 48B rows, b128-aligned, conflict-free
#define VP    264     // v row pitch in halves ([d][pos])
#define PP    264     // P row pitch in halves ([m][n])

// LDS layout (bytes): P 135168 | qT 12288 | kT 12288 | v 8448 | rowmax 1024 | sinv 1024
#define ATTN_LDS_BYTES (256*PP*2 + 256*QP*2 + 256*QP*2 + 16*VP*2 + 256*4 + 256*4)

// ---------------------------------------------------------------- K1: LN stats
__global__ __launch_bounds__(256) void ln_stats_kernel(
    const float* __restrict__ lms, const float* __restrict__ pan,
    float* __restrict__ stats)  // [pan_mean | pan_rstd | lms_mean | lms_rstd] x 65536
{
  int p = blockIdx.x * 256 + threadIdx.x;
  float s0 = 0.f, q0 = 0.f, s1 = 0.f, q1 = 0.f;
  #pragma unroll
  for (int c = 0; c < 32; ++c) {
    float a = pan[(size_t)c * HW + p]; s0 += a; q0 += a * a;
    float b = lms[(size_t)c * HW + p]; s1 += b; q1 += b * b;
  }
  float m0 = s0 * (1.f / 32.f), v0 = q0 * (1.f / 32.f) - m0 * m0;
  float m1 = s1 * (1.f / 32.f), v1 = q1 * (1.f / 32.f) - m1 * m1;
  stats[p]            = m0;
  stats[HW + p]       = rsqrtf(v0 + 1e-5f);
  stats[2 * HW + p]   = m1;
  stats[3 * HW + p]   = rsqrtf(v1 + 1e-5f);
}

// ------------------------------------------------- K2: fused LN + 1x1 conv
// y[dc,p] = rstd[p]*(sum_c Weff[dc,c]*x[c,p] - mean[p]*sum_c Weff[dc,c]) + b[dc]
// with Weff[dc,c] = W[dc,c]*g[c].
__global__ __launch_bounds__(256) void conv1x1_ln_kernel(
    const float* __restrict__ x, const float* __restrict__ mean,
    const float* __restrict__ rstd, const float* __restrict__ g,
    const float* __restrict__ w, const float* __restrict__ bias,
    _Float16* __restrict__ y, int Cout)
{
  __shared__ float Wl[256 * 32];
  __shared__ float Bl[256];
  int t = threadIdx.x;
  for (int i = t; i < Cout * 32; i += 256) Wl[i] = w[i] * g[i & 31];
  if (t < Cout) Bl[t] = bias[t];
  __syncthreads();

  int p = blockIdx.x * 256 + t;
  float xv[32];
  #pragma unroll
  for (int c = 0; c < 32; ++c) xv[c] = x[(size_t)c * HW + p];
  float mu = mean[p], rs = rstd[p];
  for (int dc = 0; dc < Cout; ++dc) {
    const float* wr = &Wl[dc * 32];
    float acc = 0.f, wsum = 0.f;
    #pragma unroll
    for (int c = 0; c < 32; ++c) { acc += wr[c] * xv[c]; wsum += wr[c]; }
    y[(size_t)dc * HW + p] = (_Float16)((acc - mu * wsum) * rs + Bl[dc]);
  }
}

// ---------------------------------------------------- K3: depthwise 3x3 SAME
__global__ __launch_bounds__(256) void dw3x3_kernel(
    const _Float16* __restrict__ x, const float* __restrict__ w,
    const float* __restrict__ bias, _Float16* __restrict__ y)
{
  int bx = blockIdx.x;
  int c = bx >> 8;                       // 256 blocks per channel
  int p = ((bx & 255) << 8) + threadIdx.x;
  int y0 = p >> 8, x0 = p & 255;
  const float* wc = &w[c * 9];
  const _Float16* xc = x + (size_t)c * HW;
  float acc = bias[c];
  #pragma unroll
  for (int dy = -1; dy <= 1; ++dy) {
    int yy = y0 + dy;
    if ((unsigned)yy >= 256u) continue;
    #pragma unroll
    for (int dx = -1; dx <= 1; ++dx) {
      int xx = x0 + dx;
      if ((unsigned)xx >= 256u) continue;
      acc += wc[(dy + 1) * 3 + (dx + 1)] * (float)xc[yy * 256 + xx];
    }
  }
  y[(size_t)c * HW + p] = (_Float16)acc;
}

// --------------------------------------------------------- K4: attention
__global__ __launch_bounds__(256) void attn_kernel(
    const _Float16* __restrict__ Q2, const _Float16* __restrict__ KV2,
    float* __restrict__ attn_out, float* __restrict__ out_out)
{
  extern __shared__ char smem[];
  _Float16* P  = (_Float16*)smem;          // [256][PP] scores -> probs (f16)
  _Float16* qT = P + 256 * PP;             // [m][d]  (pitch QP)
  _Float16* kT = qT + 256 * QP;            // [n][d]  (pitch QP), k normalized
  _Float16* vS = kT + 256 * QP;            // [d][n]  (pitch VP)
  float* rowmax = (float*)(vS + 16 * VP);  // [256]
  float* sinv   = rowmax + 256;            // [256]

  const int t    = threadIdx.x;
  const int lane = t & 31;
  const int wave = t >> 5;
  const int head = blockIdx.x & 7;
  const int win  = blockIdx.x >> 3;
  const int wy = win >> 4, wx = win & 15;
  const int wbase = wy * 256 + wx;         // pixel(m) = (m>>4)*4096 + (m&15)*16 + wbase

  // ---- load q (thread = m) into qT[m][d] ----
  {
    int m = t, iy = m >> 4, ix = m & 15;
    int pix = iy * 4096 + ix * 16 + wbase;
    const _Float16* qp = Q2 + (size_t)(head * 16) * HW + pix;
    #pragma unroll
    for (int d = 0; d < 16; ++d) qT[m * QP + d] = qp[(size_t)d * HW];
  }
  // ---- load k (L2-normalize over the 256 window positions) and v ----
  {
    int d = t >> 4, j = t & 15;            // lanes of equal d are one 16-lane group
    const _Float16* kp = KV2 + (size_t)(head * 16 + d) * HW;
    const _Float16* vp = KV2 + (size_t)(128 + head * 16 + d) * HW;
    float kv[16];
    float ss = 0.f;
    #pragma unroll
    for (int i = 0; i < 16; ++i) {
      int m = j * 16 + i, iy = m >> 4, ix = m & 15;
      int pix = iy * 4096 + ix * 16 + wbase;
      float kl = (float)kp[pix];
      kv[i] = kl; ss += kl * kl;
      vS[d * VP + m] = vp[pix];
    }
    ss += __shfl_xor(ss, 1);  ss += __shfl_xor(ss, 2);
    ss += __shfl_xor(ss, 4);  ss += __shfl_xor(ss, 8);
    float rinv = 1.0f / fmaxf(sqrtf(ss), 1e-12f);
    #pragma unroll
    for (int i = 0; i < 16; ++i)
      kT[(j * 16 + i) * QP + d] = (_Float16)(kv[i] * rinv);
  }
  __syncthreads();

  // ---- GEMM1: S[m,n] = sum_d q[d,m]*k[d,n]  (M=N=16 tiles, K=16 padded to 32)
  // Wave w owns m-tiles {2w, 2w+1} across all 16 n-tiles.
  {
    const int l15   = lane & 15;
    const int kAoff = (lane < 16) ? 0 : 8;   // A frag elems0-7 -> K base
    #pragma unroll
    for (int tr = 0; tr < 2; ++tr) {
      int tm = wave * 2 + tr;
      V16 a;
      a.h[0] = *(const v8h*)(qT + (tm * 16 + l15) * QP + kAoff);
      a.h[1] = v8h_zero();                   // K = 16..31 padding
      int mrow0 = tm * 16 + ((lane >> 4) << 3);
      for (int tn = 0; tn < 16; ++tn) {
        V16 b;                               // B: elems0-7 hold K kAoff..kAoff+7, elems8-15 hold K 16..31 (=0)
        b.h[0] = *(const v8h*)(kT + (tn * 16 + l15) * QP + kAoff);
        b.h[1] = v8h_zero();
        v8f c = v8f_zero();
        c = __builtin_amdgcn_wmma_f32_16x16x32_f16(false, a.v, false, b.v,
                                                   (short)0, c, false, false);
        int ncol = tn * 16 + l15;
        #pragma unroll
        for (int r = 0; r < 8; ++r)
          P[(mrow0 + r) * PP + ncol] = (_Float16)c[r];
      }
    }
  }
  __syncthreads();

  // ---- softmax pass 1: row max + row sum (thread = row) ----
  {
    float mx = -3.0e38f;
    for (int n = 0; n < 256; ++n)
      mx = fmaxf(mx, (float)P[t * PP + n]);
    float s = 0.f;
    for (int n = 0; n < 256; ++n)
      s += __expf((float)P[t * PP + n] - mx);
    rowmax[t] = mx;
    sinv[t]   = 1.0f / s;
  }
  __syncthreads();

  // ---- softmax pass 2: normalize, write attn (coalesced: thread = n) ----
  {
    size_t abase = ((size_t)(win * 8 + head)) << 16;
    for (int i = 0; i < 256; ++i) {
      float sv = (float)P[i * PP + t];
      float p  = __expf(sv - rowmax[i]) * sinv[i];
      attn_out[abase + ((size_t)i << 8) + t] = p;
      P[i * PP + t] = (_Float16)p;
    }
  }
  __syncthreads();

  // ---- GEMM2: out[m,d] = sum_n P[m,n] * v[d,n]   (K = 256, 8 WMMA steps)
  {
    const int l15 = lane & 15;
    const int nbA = (lane < 16) ? 0 : 8;
    const int nbB = (lane < 16) ? 0 : 8;
    #pragma unroll
    for (int tr = 0; tr < 2; ++tr) {
      int tm = wave * 2 + tr;
      v8f c = v8f_zero();
      for (int kb = 0; kb < 8; ++kb) {
        V16 a, b;
        const _Float16* pa = P + (tm * 16 + l15) * PP + kb * 32 + nbA;
        a.h[0] = *(const v8h*)(pa);
        a.h[1] = *(const v8h*)(pa + 16);
        const _Float16* pb = vS + l15 * VP + kb * 32 + nbB;
        b.h[0] = *(const v8h*)(pb);
        b.h[1] = *(const v8h*)(pb + 16);
        c = __builtin_amdgcn_wmma_f32_16x16x32_f16(false, a.v, false, b.v,
                                                   (short)0, c, false, false);
      }
      int mrow0 = tm * 16 + ((lane >> 4) << 3);
      float* op = out_out + (size_t)(head * 16 + l15) * HW + wbase;
      #pragma unroll
      for (int r = 0; r < 8; ++r) {
        int m = mrow0 + r, iy = m >> 4, ix = m & 15;
        op[iy * 4096 + ix * 16] = c[r];
      }
    }
  }
}

// ---------------------------------------------------------------------------
extern "C" void kernel_launch(void* const* d_in, const int* in_sizes, int n_in,
                              void* d_out, int out_size, void* d_ws, size_t ws_size,
                              hipStream_t stream)
{
  (void)in_sizes; (void)n_in; (void)out_size; (void)ws_size;
  const float* lms   = (const float*)d_in[0];
  const float* pan   = (const float*)d_in[1];
  const float* g_ms  = (const float*)d_in[2];
  const float* g_pan = (const float*)d_in[3];
  const float* q_w1  = (const float*)d_in[4];
  const float* q_b1  = (const float*)d_in[5];
  const float* q_w2  = (const float*)d_in[6];
  const float* q_b2  = (const float*)d_in[7];
  const float* kv_w1 = (const float*)d_in[8];
  const float* kv_b1 = (const float*)d_in[9];
  const float* kv_w2 = (const float*)d_in[10];
  const float* kv_b2 = (const float*)d_in[11];

  // workspace layout (bytes):
  //  [0,1MB)      LN stats (4 x 65536 f32)
  //  [1MB, +16MB) Q1  f16  128 x 65536
  //  [.., +32MB)  KV1 f16  256 x 65536
  //  [.., +16MB)  Q2  f16
  //  [.., +32MB)  KV2 f16          (total ~97 MB)
  char* ws = (char*)d_ws;
  float*    stats = (float*)ws;
  _Float16* Q1  = (_Float16*)(ws + (size_t)(1u << 20));
  _Float16* KV1 = (_Float16*)(ws + (size_t)(1u << 20) + 16777216u);
  _Float16* Q2  = (_Float16*)(ws + (size_t)(1u << 20) + 50331648u);
  _Float16* KV2 = (_Float16*)(ws + (size_t)(1u << 20) + 67108864u);

  float* attn = (float*)d_out;                       // 256*8*256*256 f32
  float* outp = attn + (size_t)134217728;            // 128*256*256 f32

  ln_stats_kernel<<<256, 256, 0, stream>>>(lms, pan, stats);
  conv1x1_ln_kernel<<<256, 256, 0, stream>>>(pan, stats, stats + HW, g_pan,
                                             q_w1, q_b1, Q1, 128);
  conv1x1_ln_kernel<<<256, 256, 0, stream>>>(lms, stats + 2 * HW, stats + 3 * HW, g_ms,
                                             kv_w1, kv_b1, KV1, 256);
  dw3x3_kernel<<<128 * 256, 256, 0, stream>>>(Q1, q_w2, q_b2, Q2);
  dw3x3_kernel<<<256 * 256, 256, 0, stream>>>(KV1, kv_w2, kv_b2, KV2);

  (void)hipFuncSetAttribute((const void*)attn_kernel,
                            hipFuncAttributeMaxDynamicSharedMemorySize,
                            (int)ATTN_LDS_BYTES);
  attn_kernel<<<2048, 256, ATTN_LDS_BYTES, stream>>>(Q2, KV2, attn, outp);
}
